// BahdanauAttention_74225624809808
// MI455X (gfx1250) — matrix operand order repaired and checked by reading, not compiled, not run on hardware
//
#include <hip/hip_runtime.h>
#include <hip/hip_bf16.h>

#define BB 16
#define SQl 128
#define SVl 128
#define DQl 512
#define DVl 512
#define UU  256
#define QT  4          // query rows per score block

typedef __attribute__((ext_vector_type(16))) _Float16 v16h;
typedef __attribute__((ext_vector_type(8)))  float    v8f;

// ---------------------------------------------------------------------------
// gfx1250 hardware tanh (V_TANH_F32). Builtin if present, else inline asm.
// v_nop after the trans op covers the TRANS32 RAW hazard (compiler cannot
// schedule around opaque asm).
// ---------------------------------------------------------------------------
__device__ __forceinline__ float tanh_hw(float x) {
#if __has_builtin(__builtin_amdgcn_tanhf)
  return __builtin_amdgcn_tanhf(x);
#elif __has_builtin(__builtin_amdgcn_tanh_f32)
  return __builtin_amdgcn_tanh_f32(x);
#else
  float y;
  asm volatile("v_tanh_f32 %0, %1\n\tv_nop" : "=v"(y) : "v"(x));
  return y;
#endif
}

// ---------------------------------------------------------------------------
// One wave32 computes one 16x16 f32 tile of C = A * B where A,B are f32
// row-major in memory; fragments are converted to f16 and accumulated with
// v_wmma_f32_16x16x32_f16 (f32 accumulator).
//
// Wave32 fragment layouts used here (A/B in contiguous k order, not the ISA table order):
//   A (16xK f16): lane holds row M=lane%16; elem e -> K=k0+half*16+e
//                 (half = lane/16)
//   B (Kx16 f16): lane holds col N=lane%16; elem e -> K=k0+half*16+e
//   C/D (16x16 f32, 8 VGPRs): elem r -> row M=half*8+r, col N=lane%16
// ---------------------------------------------------------------------------
__device__ __forceinline__ v8f wmma_tile_f16(const float* __restrict__ A, int lda,
                                             const float* __restrict__ Bm, int ldb,
                                             int K) {
  const int lane = threadIdx.x & 31;
  const int half = lane >> 4;
  const int lm   = lane & 15;
  v8f acc = {};
  for (int k0 = 0; k0 < K; k0 += 32) {
    v16h a, b;
    const float* arow = A + (size_t)lm * lda + k0 + half * 16;
#pragma unroll
    for (int e = 0; e < 8; ++e) {
      a[e]     = (_Float16)arow[e];
      a[e + 8] = (_Float16)arow[e + 8];
    }
    const float* bcol = Bm + (size_t)(k0 + half * 16) * ldb + lm;
#pragma unroll
    for (int e = 0; e < 16; ++e) b[e] = (_Float16)bcol[(size_t)e * ldb];
    acc = __builtin_amdgcn_wmma_f32_16x16x32_f16(false, a, false, b,
                                                 (short)0, acc, false, false);
  }
  return acc;
}

__device__ __forceinline__ void store_tile(float* __restrict__ C, int ldc,
                                           v8f acc, float cbias) {
  const int lane = threadIdx.x & 31;
  const int half = lane >> 4;
  const int lm   = lane & 15;
  float* cp = C + (size_t)(half * 8) * ldc + lm;
#pragma unroll
  for (int r = 0; r < 8; ++r) cp[(size_t)r * ldc] = acc[r] + cbias;
}

// ---------------------------------------------------------------------------
// Kernel 1: Out[M,N] = X[M,K] @ W[K,N] + bias[N]   (s1 / s2 projections)
// One wave per 16x16 tile, 4 waves (128 threads) per block.
// ---------------------------------------------------------------------------
__global__ void sgemm_bias_wmma(const float* __restrict__ X,
                                const float* __restrict__ W,
                                const float* __restrict__ bias,
                                float* __restrict__ Out,
                                int M, int K, int N) {
  const int wave  = threadIdx.x >> 5;
  const int tile  = blockIdx.x * (blockDim.x >> 5) + wave;
  const int ntile = N >> 4;
  const int mTile = tile / ntile;
  const int nTile = tile % ntile;
  if (mTile >= (M >> 4)) return;  // wave-uniform guard; EXEC all-ones inside

  const float* A  = X + (size_t)(mTile * 16) * K;
  const float* Bm = W + nTile * 16;
  v8f acc = wmma_tile_f16(A, K, Bm, N, K);
  const float cb = bias[nTile * 16 + (threadIdx.x & 15)];
  store_tile(Out + (size_t)(mTile * 16) * N + nTile * 16, N, acc, cb);
}

// ---------------------------------------------------------------------------
// Kernel 2: fused score + softmax, q-tiled and b128-vectorized.
// One block handles QT=4 query rows of one batch. Thread t owns v=t.
// Inner loop per 4 u's: 1 global b128 (s2, reused across QT rows),
// QT ds b128 (s1 rows), 1 ds b128 (Vw), then 4*QT tanh+fma.
// Never materializes the [B,Sq,Sv,U] tensor (268 MB avoided).
// ---------------------------------------------------------------------------
__global__ void score_softmax_kernel(const float* __restrict__ s1,
                                     const float* __restrict__ s2,
                                     const float* __restrict__ Vw,
                                     const float* __restrict__ Vb,
                                     float* __restrict__ attn) {
  __shared__ __align__(16) float sh1[QT * UU];
  __shared__ __align__(16) float shw[UU];
  __shared__ float red[SVl];

  const int t   = threadIdx.x;                 // 0..127 : one v per thread
  const int blk = blockIdx.x;                  // 0 .. B*Sq/QT - 1
  const int b   = blk / (SQl / QT);
  const int q0  = (blk % (SQl / QT)) * QT;

  // QT consecutive s1 rows are contiguous: flat copy into LDS
  const float* s1base = s1 + ((size_t)(b * SQl + q0)) * UU;
#pragma unroll
  for (int i = t; i < QT * UU; i += SVl) sh1[i] = s1base[i];
  shw[t]       = Vw[t];
  shw[t + 128] = Vw[t + 128];
  __syncthreads();

  const float4* s2row = (const float4*)(s2 + ((size_t)(b * SVl + t)) * UU);
  const float4* w4    = (const float4*)shw;

  float sc[QT];
#pragma unroll
  for (int qi = 0; qi < QT; ++qi) sc[qi] = 0.f;

#pragma unroll 2
  for (int u4 = 0; u4 < UU / 4; ++u4) {
    const float4 x = s2row[u4];                // 1 global b128, reused QT times
    const float4 w = w4[u4];                   // 1 ds b128
#pragma unroll
    for (int qi = 0; qi < QT; ++qi) {
      const float4 a = ((const float4*)(sh1 + qi * UU))[u4];  // ds b128
      sc[qi] = fmaf(w.x, tanh_hw(a.x + x.x), sc[qi]);
      sc[qi] = fmaf(w.y, tanh_hw(a.y + x.y), sc[qi]);
      sc[qi] = fmaf(w.z, tanh_hw(a.z + x.z), sc[qi]);
      sc[qi] = fmaf(w.w, tanh_hw(a.w + x.w), sc[qi]);
    }
  }
  const float vb = Vb[0];
#pragma unroll
  for (int qi = 0; qi < QT; ++qi) sc[qi] += vb;

  // block-wide softmax over Sv=128 for each of the QT rows
#pragma unroll 1
  for (int qi = 0; qi < QT; ++qi) {
    red[t] = sc[qi];
    __syncthreads();
    for (int s = 64; s > 0; s >>= 1) {
      if (t < s) red[t] = fmaxf(red[t], red[t + s]);
      __syncthreads();
    }
    const float m = red[0];
    __syncthreads();
    const float e = __expf(sc[qi] - m);
    red[t] = e;
    __syncthreads();
    for (int s = 64; s > 0; s >>= 1) {
      if (t < s) red[t] += red[t + s];
      __syncthreads();
    }
    attn[((size_t)(b * SQl + q0 + qi)) * SVl + t] = e / red[0];
    __syncthreads();   // red reused next iteration
  }
}

// ---------------------------------------------------------------------------
// Kernel 3: context[b] = attn[b] (Sq x Sv) @ values[b] (Sv x Dv), WMMA tiles.
// ---------------------------------------------------------------------------
__global__ void context_wmma(const float* __restrict__ attn,
                             const float* __restrict__ values,
                             float* __restrict__ ctx) {
  const int wave = threadIdx.x >> 5;
  const int tile = blockIdx.x * (blockDim.x >> 5) + wave; // 0..4095
  const int b     = tile >> 8;        // 8 mTiles * 32 nTiles per batch
  const int mTile = (tile >> 5) & 7;
  const int nTile = tile & 31;

  const float* A  = attn   + (size_t)b * SQl * SVl + (size_t)(mTile * 16) * SVl;
  const float* Bm = values + (size_t)b * SVl * DVl + nTile * 16;
  v8f acc = wmma_tile_f16(A, SVl, Bm, DVl, SVl);
  store_tile(ctx + (size_t)b * SQl * DVl + (size_t)(mTile * 16) * DVl + nTile * 16,
             DVl, acc, 0.f);
}

// ---------------------------------------------------------------------------
extern "C" void kernel_launch(void* const* d_in, const int* in_sizes, int n_in,
                              void* d_out, int out_size, void* d_ws, size_t ws_size,
                              hipStream_t stream) {
  const float* query  = (const float*)d_in[0];
  const float* values = (const float*)d_in[1];
  const float* W1     = (const float*)d_in[2];
  const float* b1     = (const float*)d_in[3];
  const float* W2     = (const float*)d_in[4];
  const float* b2     = (const float*)d_in[5];
  const float* Vw     = (const float*)d_in[6];
  const float* Vb     = (const float*)d_in[7];

  float* ctx  = (float*)d_out;                       // [B,Sq,Dv]
  float* attn = ctx + (size_t)BB * SQl * DVl;        // [B,Sq,Sv] (==[...,1] flat)

  float* s1 = (float*)d_ws;                          // [B*Sq, U]  2 MB
  float* s2 = s1 + (size_t)BB * SQl * UU;            // [B*Sv, U]  2 MB

  // s1 = query@W1+b1 : M=2048,K=512,N=256 -> 2048 tiles / 4 waves = 512 blocks
  sgemm_bias_wmma<<<512, 128, 0, stream>>>(query,  W1, b1, s1, BB * SQl, DQl, UU);
  // s2 = values@W2+b2
  sgemm_bias_wmma<<<512, 128, 0, stream>>>(values, W2, b2, s2, BB * SVl, DVl, UU);
  // fused tanh-score + softmax -> attention weights (also an output)
  score_softmax_kernel<<<(BB * SQl) / QT, 128, 0, stream>>>(s1, s2, Vw, Vb, attn);
  // context = attn @ values : 16 batches * 8 * 32 tiles = 4096 / 4 = 1024 blocks
  context_wmma<<<1024, 128, 0, stream>>>(attn, values, ctx);
}
